// CriticPointNet2_61967788147087
// MI455X (gfx1250) — compile-verified, matrix operand order checked by reading, not run on hardware
//
#include <hip/hip_runtime.h>
#include <hip/hip_bf16.h>

// PointNet++ critic forward for MI455X (gfx1250, wave32).
// All einsum/matmul layers run on v_wmma_f32_16x16x32_f16 (f16 in, f32 acc).
// Weights converted once to f16 with K padded to mult-of-32, Cout to mult-of-16.
// sa1 stages its final-layer weight into LDS via the Tensor Data Mover
// (tensor_load_to_lds + s_wait_tensorcnt), overlapped with layers 0/1.

typedef __attribute__((ext_vector_type(16))) _Float16 v16h;
typedef __attribute__((ext_vector_type(8)))  float    v8f;
typedef __attribute__((ext_vector_type(4)))  unsigned u32x4;
typedef __attribute__((ext_vector_type(8)))  unsigned u32x8;

#define DEV __device__ __forceinline__

// ---------------------------------------------------------------------------
// Tensor Data Mover: DMA `ndw` dwords from global `gptr` into LDS `lds_off`.
// 2D descriptor with a single row (data_size = 4B). Issue from ONE wave only
// (EXEC is ignored; TENSORcnt is per-wave). Inline asm per CDNA5_HIP.md since
// the clang builtin arity differs across toolchains.
// ---------------------------------------------------------------------------
DEV void tdm_load_to_lds(unsigned lds_off, const void* gptr, unsigned ndw) {
  const unsigned long long ga = (unsigned long long)(size_t)gptr;
  u32x4 g0;
  g0[0] = 1u;                                        // count=1, user mode
  g0[1] = lds_off;                                   // LDS byte address
  g0[2] = (unsigned)ga;                              // global_addr[31:0]
  g0[3] = (unsigned)((ga >> 32) & 0x01FFFFFFu)       // global_addr[56:32]
          | 0x80000000u;                             // type=2 ("image")
  u32x8 g1;
  g1[0] = (2u << 16);                                // data_size=4B, mask=0
  g1[1] = (ndw & 0xFFFFu) << 16;                     // tensor_dim0[15:0]
  g1[2] = ((ndw >> 16) & 0xFFFFu) | (1u << 16);      // dim0[31:16], dim1=1
  g1[3] = (ndw & 0xFFFFu) << 16;                     // tile_dim0 (<=65535)
  g1[4] = 1u;                                        // tile_dim1=1, tile_dim2=0
  g1[5] = ndw;                                       // tensor_dim0_stride[31:0]
  g1[6] = 0u;
  g1[7] = 0u;
  asm volatile("tensor_load_to_lds %0, %1" :: "s"(g0), "s"(g1) : "memory");
}

// ---------------------------------------------------------------------------
// WMMA fragment loader.
// A (16x32 f16, MxK): lanes 0-15 hold row M=lane, K in {0..7,16..23};
// lanes 16-31 hold row M=lane-16, K in {8..15,24..31}; pairs per VGPR.
// B (32x16, KxN) mirrors A with N in place of M (we compute Y = X * W^T, so
// B[k][n] = W[n][k] and a row-major W[Cout][Kpad] gives contiguous pairs).
// ---------------------------------------------------------------------------
DEV v16h load_frag16(const _Float16* P, int ld, int r0, int k0, int lane) {
  const int r = r0 + (lane & 15);
  const _Float16* row = P + (size_t)r * ld + k0 + ((lane >> 4) << 3);
  v16h f;
#pragma unroll
  for (int v = 0; v < 4; ++v) {
    f[2 * v]     = row[2 * v];
    f[2 * v + 1] = row[2 * v + 1];
    f[8 + 2 * v]     = row[16 + 2 * v];
    f[8 + 2 * v + 1] = row[16 + 2 * v + 1];
  }
  return f;
}

// One MLP layer: Y[M][Cout] = relu(X[M][Kpad] * W[Cout][Kpad]^T + bias), f16 out.
// C/D layout: VGPR r -> row (r + 8*half), col = lane&15.
DEV void mlp_layer(const _Float16* Xin, int ldi, int M, int Kpad,
                   const _Float16* W, const float* bias, int Cout,
                   _Float16* Xout, int ldo, int wave, int nwaves, int lane) {
  const int mtiles = M >> 4, ntiles = Cout >> 4, ktiles = Kpad >> 5;
  const int nl = lane & 15, half = lane >> 4;
  for (int t = wave; t < mtiles * ntiles; t += nwaves) {
    const int mt = t / ntiles, nt = t - mt * ntiles;
    v8f c = {};
    for (int kt = 0; kt < ktiles; ++kt) {
      if (kt + 1 < ktiles)   // prefetch next K-slice of W (global_prefetch_b8)
        __builtin_prefetch(W + (size_t)(nt * 16 + nl) * Kpad + (kt + 1) * 32, 0, 1);
      v16h a = load_frag16(Xin, ldi, mt * 16, kt * 32, lane);
      v16h b = load_frag16(W,  Kpad, nt * 16, kt * 32, lane);
      c = __builtin_amdgcn_wmma_f32_16x16x32_f16(false, a, false, b,
                                                 (short)0, c, false, false);
    }
    const float bv = bias[nt * 16 + nl];
#pragma unroll
    for (int r = 0; r < 8; ++r) {
      float y = c[r] + bv;
      y = y > 0.f ? y : 0.f;
      Xout[(size_t)(mt * 16 + r + 8 * half) * ldo + nt * 16 + nl] = (_Float16)y;
    }
  }
}

// Final layer fused with max-pool over points (rows). relu >= 0 so init 0.
template <typename TO>
DEV void mlp_maxout(const _Float16* Xin, int ldi, int M, int Kpad,
                    const _Float16* W, const float* bias, int Cout,
                    TO* out, int wave, int nwaves, int lane) {
  const int mtiles = M >> 4, ntiles = Cout >> 4, ktiles = Kpad >> 5;
  const int nl = lane & 15, half = lane >> 4;
  for (int nt = wave; nt < ntiles; nt += nwaves) {
    const float bv = bias[nt * 16 + nl];
    float vmax = 0.f;
    for (int mt = 0; mt < mtiles; ++mt) {
      v8f c = {};
      for (int kt = 0; kt < ktiles; ++kt) {
        v16h a = load_frag16(Xin, ldi, mt * 16, kt * 32, lane);
        v16h b = load_frag16(W,  Kpad, nt * 16, kt * 32, lane);
        c = __builtin_amdgcn_wmma_f32_16x16x32_f16(false, a, false, b,
                                                   (short)0, c, false, false);
      }
#pragma unroll
      for (int r = 0; r < 8; ++r) {
        float y = c[r] + bv;
        y = y > 0.f ? y : 0.f;
        vmax = vmax > y ? vmax : y;
      }
    }
    vmax = fmaxf(vmax, __shfl_xor(vmax, 16, 32));   // merge the two row halves
    if (half == 0) out[nt * 16 + nl] = (TO)vmax;
  }
}

// ---------------------------------------------------------------------------
// Data-prep kernels
// ---------------------------------------------------------------------------
__global__ void transpose_kernel(const float* __restrict__ xyz,
                                 float* __restrict__ pts) {
  const int g = blockIdx.x * blockDim.x + threadIdx.x;   // b*4096 + n
  if (g >= 32 * 4096) return;
  const int b = g >> 12, n = g & 4095;
#pragma unroll
  for (int c = 0; c < 3; ++c)
    pts[(size_t)g * 3 + c] = xyz[((size_t)b * 3 + c) * 4096 + n];
}

__global__ void convw_kernel(const float* __restrict__ src, _Float16* __restrict__ dst,
                             int O, int C, int Op, int Cp) {
  const int g = blockIdx.x * blockDim.x + threadIdx.x;
  if (g >= Op * Cp) return;
  const int o = g / Cp, c = g - o * Cp;
  dst[g] = (o < O && c < C) ? (_Float16)src[(size_t)o * C + c] : (_Float16)0.f;
}

// ---------------------------------------------------------------------------
// Farthest point sampling: one block per batch, matches jax.lax.scan semantics
// (records 'far' before the min-update; idx[0] is always 0).
// ---------------------------------------------------------------------------
__global__ void fps_kernel(const float* __restrict__ xyz, int N, int npoint,
                           int* __restrict__ outIdx, float* __restrict__ newXyz) {
  extern __shared__ unsigned char fps_sm[];
  float* dist = (float*)fps_sm;           // [N]
  float* sval = dist + N;                 // [T]
  int*   sidx = (int*)(sval + blockDim.x);
  const int b = blockIdx.x, tid = threadIdx.x, T = blockDim.x;
  const float* px = xyz + (size_t)b * N * 3;
  for (int j = tid; j < N; j += T) dist[j] = 1e10f;
  __syncthreads();
  int far = 0;
  for (int it = 0; it < npoint; ++it) {
    if (tid == 0) {
      outIdx[(size_t)b * npoint + it] = far;
      newXyz[((size_t)b * npoint + it) * 3 + 0] = px[far * 3 + 0];
      newXyz[((size_t)b * npoint + it) * 3 + 1] = px[far * 3 + 1];
      newXyz[((size_t)b * npoint + it) * 3 + 2] = px[far * 3 + 2];
    }
    const float cx = px[far * 3 + 0], cy = px[far * 3 + 1], cz = px[far * 3 + 2];
    float bv = -1.f; int bi = 0;
    for (int j = tid; j < N; j += T) {
      const float dx = px[j * 3 + 0] - cx, dy = px[j * 3 + 1] - cy,
                  dz = px[j * 3 + 2] - cz;
      const float d2 = dx * dx + dy * dy + dz * dz;
      float dm = dist[j];
      dm = d2 < dm ? d2 : dm;
      dist[j] = dm;
      if (dm > bv) { bv = dm; bi = j; }
    }
    sval[tid] = bv; sidx[tid] = bi;
    __syncthreads();
    for (int o = T >> 1; o > 0; o >>= 1) {
      if (tid < o) {
        const float v2 = sval[tid + o]; const int i2 = sidx[tid + o];
        if (v2 > sval[tid] || (v2 == sval[tid] && i2 < sidx[tid])) {
          sval[tid] = v2; sidx[tid] = i2;
        }
      }
      __syncthreads();
    }
    far = sidx[0];
    __syncthreads();
  }
}

// ---------------------------------------------------------------------------
// Ball query: first nsample in-radius indices in ascending order, pad w/ first
// (equivalent to reference's mask/sort/truncate).
// ---------------------------------------------------------------------------
__global__ void ballq_kernel(const float* __restrict__ xyz, const float* __restrict__ nxyz,
                             int N, int S, int nsample, float r2, int total,
                             int* __restrict__ gidx) {
  const int gid = blockIdx.x * blockDim.x + threadIdx.x;   // b*S + s
  if (gid >= total) return;
  const int b = gid / S;
  const float* px = xyz + (size_t)b * N * 3;
  const float cx = nxyz[(size_t)gid * 3 + 0], cy = nxyz[(size_t)gid * 3 + 1],
              cz = nxyz[(size_t)gid * 3 + 2];
  int* g = gidx + (size_t)gid * nsample;
  int cnt = 0, first = 0;
  for (int j = 0; j < N && cnt < nsample; ++j) {
    if ((j & 63) == 0 && j + 64 < N)
      __builtin_prefetch(px + (size_t)(j + 64) * 3, 0, 1);
    const float dx = px[j * 3 + 0] - cx, dy = px[j * 3 + 1] - cy,
                dz = px[j * 3 + 2] - cz;
    if (dx * dx + dy * dy + dz * dz <= r2) {
      if (cnt == 0) first = j;
      g[cnt++] = j;
    }
  }
  for (int k = cnt; k < nsample; ++k) g[k] = first;
}

// ---------------------------------------------------------------------------
// Stage 1: one block per (b,s) group of 32 points. 3->64->64->128 + max pool.
// W2 (128x64 f16, 16 KB) is DMA'd into LDS by the TDM, overlapped w/ L0/L1.
// ---------------------------------------------------------------------------
__global__ void __launch_bounds__(128) sa1_kernel(
    const float* __restrict__ pts, const float* __restrict__ nx1,
    const int* __restrict__ gidx,
    const _Float16* __restrict__ W0, const float* __restrict__ b0,
    const _Float16* __restrict__ W1, const float* __restrict__ b1,
    const _Float16* __restrict__ W2, const float* __restrict__ b2,
    _Float16* __restrict__ l1p) {
  __shared__ _Float16 bufA[32 * 64];
  __shared__ _Float16 bufB[32 * 64];
  __shared__ _Float16 bufW2[128 * 64];              // TDM staging, 16 KB
  const int bs = blockIdx.x, b = bs >> 9;           // S = 512
  const int tid = threadIdx.x, lane = tid & 31, wave = tid >> 5;
  // Kick off the async weight DMA from one wave (EXEC ignored by TDM).
  if (wave == 0)
    tdm_load_to_lds((unsigned)(size_t)(void*)bufW2, W2, (128 * 64 * 2) / 4);
  const int* gi = gidx + (size_t)bs * 32;
  const float cx = nx1[(size_t)bs * 3 + 0], cy = nx1[(size_t)bs * 3 + 1],
              cz = nx1[(size_t)bs * 3 + 2];
  if (tid < 32) {
    const int idx = gi[tid];
    const float* pp = pts + ((size_t)b * 4096 + idx) * 3;
    _Float16* row = bufA + tid * 32;
    row[0] = (_Float16)(pp[0] - cx);
    row[1] = (_Float16)(pp[1] - cy);
    row[2] = (_Float16)(pp[2] - cz);
    for (int k = 3; k < 32; ++k) row[k] = (_Float16)0.f;
  }
  __syncthreads();
  mlp_layer(bufA, 32, 32, 32, W0, b0, 64, bufB, 64, wave, 4, lane);
  __syncthreads();
  mlp_layer(bufB, 64, 32, 64, W1, b1, 64, bufA, 64, wave, 4, lane);
  if (wave == 0) __builtin_amdgcn_s_wait_tensorcnt(0);  // DMA done
  __syncthreads();
  mlp_maxout<_Float16>(bufA, 64, 32, 64, bufW2, b2, 128,
                       l1p + (size_t)bs * 128, wave, 4, lane);
}

// ---------------------------------------------------------------------------
// Stage 2: one block per (b,s) group of 64 points. 131(pad160)->128->128->256.
// ---------------------------------------------------------------------------
__global__ void __launch_bounds__(128) sa2_kernel(
    const float* __restrict__ l1x, const _Float16* __restrict__ l1pf,
    const float* __restrict__ nx2, const int* __restrict__ gidx,
    const _Float16* __restrict__ W0, const float* __restrict__ b0,
    const _Float16* __restrict__ W1, const float* __restrict__ b1,
    const _Float16* __restrict__ W2, const float* __restrict__ b2,
    _Float16* __restrict__ l2p) {
  __shared__ _Float16 bufA[64 * 160];
  __shared__ _Float16 bufB[64 * 128];
  const int bs = blockIdx.x, b = bs >> 7;           // S = 128
  const int tid = threadIdx.x, lane = tid & 31, wave = tid >> 5;
  const int* gi = gidx + (size_t)bs * 64;
  const float cx = nx2[(size_t)bs * 3 + 0], cy = nx2[(size_t)bs * 3 + 1],
              cz = nx2[(size_t)bs * 3 + 2];
  for (int e = tid; e < 64 * 160; e += 128) {
    const int p = e / 160, k = e - p * 160;
    const int idx = gi[p];
    float v;
    if (k < 3) {
      const float c = (k == 0) ? cx : (k == 1) ? cy : cz;
      v = l1x[((size_t)b * 512 + idx) * 3 + k] - c;
    } else if (k < 131) {
      v = (float)l1pf[((size_t)b * 512 + idx) * 128 + (k - 3)];
    } else {
      v = 0.f;
    }
    bufA[e] = (_Float16)v;
  }
  __syncthreads();
  mlp_layer(bufA, 160, 64, 160, W0, b0, 128, bufB, 128, wave, 4, lane);
  __syncthreads();
  mlp_layer(bufB, 128, 64, 128, W1, b1, 128, bufA, 128, wave, 4, lane);
  __syncthreads();
  mlp_maxout<_Float16>(bufA, 128, 64, 128, W2, b2, 256,
                       l2p + (size_t)bs * 256, wave, 4, lane);
}

// ---------------------------------------------------------------------------
// Stage 3 (group-all) + FC head: activations staged in ws (L2-resident).
// ---------------------------------------------------------------------------
__global__ void fill3_kernel(const float* __restrict__ nx2,
                             const _Float16* __restrict__ l2p,
                             _Float16* __restrict__ X3) {
  const size_t g = (size_t)blockIdx.x * blockDim.x + threadIdx.x;
  const size_t total = (size_t)32 * 128 * 288;
  if (g >= total) return;
  const int k = (int)(g % 288);
  const size_t bp = g / 288;                        // b*128 + p
  float v;
  if (k < 3)        v = nx2[bp * 3 + k];
  else if (k < 259) v = (float)l2p[bp * 256 + (k - 3)];
  else              v = 0.f;
  X3[g] = (_Float16)v;
}

__global__ void __launch_bounds__(256) mlp_global_kernel(
    const _Float16* __restrict__ Xin, int ldi, int M, int Kpad,
    const _Float16* __restrict__ W, const float* __restrict__ bias, int Cout,
    _Float16* __restrict__ Xout, int ldo) {
  const int b = blockIdx.x;
  mlp_layer(Xin + (size_t)b * M * ldi, ldi, M, Kpad, W, bias, Cout,
            Xout + (size_t)b * M * ldo, ldo,
            threadIdx.x >> 5, blockDim.x >> 5, threadIdx.x & 31);
}

__global__ void __launch_bounds__(256) maxout_global_kernel(
    const _Float16* __restrict__ Xin, int ldi, int M, int Kpad,
    const _Float16* __restrict__ W, const float* __restrict__ bias, int Cout,
    float* __restrict__ out) {
  const int b = blockIdx.x;
  mlp_maxout<float>(Xin + (size_t)b * M * ldi, ldi, M, Kpad, W, bias, Cout,
                    out + (size_t)b * Cout,
                    threadIdx.x >> 5, blockDim.x >> 5, threadIdx.x & 31);
}

// Final 256 -> 6 (padded 16) layer: bias, no relu, writes (32,6) f32 output.
__global__ void __launch_bounds__(64) fc_head_kernel(
    const _Float16* __restrict__ Xin, const _Float16* __restrict__ W,
    const float* __restrict__ bias, float* __restrict__ out) {
  const int wave = threadIdx.x >> 5, lane = threadIdx.x & 31;
  const int mt = wave;                              // mtiles = 2, 2 waves
  v8f c = {};
  for (int kt = 0; kt < 8; ++kt) {
    v16h a  = load_frag16(Xin, 256, mt * 16, kt * 32, lane);
    v16h bf = load_frag16(W,   256, 0,       kt * 32, lane);
    c = __builtin_amdgcn_wmma_f32_16x16x32_f16(false, a, false, bf,
                                               (short)0, c, false, false);
  }
  const int n = lane & 15, half = lane >> 4;
  if (n < 6) {
    const float bv = bias[n];
#pragma unroll
    for (int r = 0; r < 8; ++r)
      out[(size_t)(mt * 16 + r + 8 * half) * 6 + n] = c[r] + bv;
  }
}

// ---------------------------------------------------------------------------
// Host orchestration
// ---------------------------------------------------------------------------
extern "C" void kernel_launch(void* const* d_in, const int* in_sizes, int n_in,
                              void* d_out, int out_size, void* d_ws, size_t ws_size,
                              hipStream_t stream) {
  (void)in_sizes; (void)n_in; (void)out_size; (void)ws_size;
  const float* xyz = (const float*)d_in[0];
  const float* s1w[3] = {(const float*)d_in[1],  (const float*)d_in[3],  (const float*)d_in[5]};
  const float* s1b[3] = {(const float*)d_in[2],  (const float*)d_in[4],  (const float*)d_in[6]};
  const float* s2w[3] = {(const float*)d_in[7],  (const float*)d_in[9],  (const float*)d_in[11]};
  const float* s2b[3] = {(const float*)d_in[8],  (const float*)d_in[10], (const float*)d_in[12]};
  const float* s3w[3] = {(const float*)d_in[13], (const float*)d_in[15], (const float*)d_in[17]};
  const float* s3b[3] = {(const float*)d_in[14], (const float*)d_in[16], (const float*)d_in[18]};
  const float* f1w = (const float*)d_in[19]; const float* f1b = (const float*)d_in[20];
  const float* f2w = (const float*)d_in[21]; const float* f2b = (const float*)d_in[22];
  const float* f3w = (const float*)d_in[23]; const float* f3b = (const float*)d_in[24];
  float* out = (float*)d_out;

  char* ws = (char*)d_ws;
  size_t off = 0;
  auto alloc = [&](size_t bytes) -> char* {
    char* p = ws + off;
    off += (bytes + 255) & ~(size_t)255;
    return p;
  };
  const int B = 32, N = 4096;
  float*    pts   = (float*)alloc((size_t)B * N * 3 * 4);
  int*      fidx1 = (int*)alloc((size_t)B * 512 * 4);
  float*    nx1   = (float*)alloc((size_t)B * 512 * 3 * 4);
  int*      gidx1 = (int*)alloc((size_t)B * 512 * 32 * 4);
  _Float16* l1p   = (_Float16*)alloc((size_t)B * 512 * 128 * 2);
  int*      fidx2 = (int*)alloc((size_t)B * 128 * 4);
  float*    nx2   = (float*)alloc((size_t)B * 128 * 3 * 4);
  int*      gidx2 = (int*)alloc((size_t)B * 128 * 64 * 4);
  _Float16* l2p   = (_Float16*)alloc((size_t)B * 128 * 256 * 2);
  _Float16* X3    = (_Float16*)alloc((size_t)B * 128 * 288 * 2);
  _Float16* a31   = (_Float16*)alloc((size_t)B * 128 * 256 * 2);
  _Float16* a32   = (_Float16*)alloc((size_t)B * 128 * 512 * 2);
  float*    feat  = (float*)alloc((size_t)B * 1024 * 4);
  _Float16* featH = (_Float16*)alloc((size_t)B * 1024 * 2);
  _Float16* fc1   = (_Float16*)alloc((size_t)B * 512 * 2);
  _Float16* fc2   = (_Float16*)alloc((size_t)B * 256 * 2);
  _Float16* ws1[3], *ws2[3], *ws3[3];
  ws1[0] = (_Float16*)alloc(64 * 32 * 2);
  ws1[1] = (_Float16*)alloc(64 * 64 * 2);
  ws1[2] = (_Float16*)alloc(128 * 64 * 2);
  ws2[0] = (_Float16*)alloc(128 * 160 * 2);
  ws2[1] = (_Float16*)alloc(128 * 128 * 2);
  ws2[2] = (_Float16*)alloc(256 * 128 * 2);
  ws3[0] = (_Float16*)alloc(256 * 288 * 2);
  ws3[1] = (_Float16*)alloc(512 * 256 * 2);
  ws3[2] = (_Float16*)alloc(1024 * 512 * 2);
  _Float16* wf1 = (_Float16*)alloc(512 * 1024 * 2);
  _Float16* wf2 = (_Float16*)alloc(256 * 512 * 2);
  _Float16* wf3 = (_Float16*)alloc(16 * 256 * 2);

  auto conv = [&](const float* src, _Float16* dst, int O, int C, int Op, int Cp) {
    const int total = Op * Cp;
    convw_kernel<<<(total + 255) / 256, 256, 0, stream>>>(src, dst, O, C, Op, Cp);
  };
  conv(s1w[0], ws1[0],   64,    3,   64,   32);
  conv(s1w[1], ws1[1],   64,   64,   64,   64);
  conv(s1w[2], ws1[2],  128,   64,  128,   64);
  conv(s2w[0], ws2[0],  128,  131,  128,  160);
  conv(s2w[1], ws2[1],  128,  128,  128,  128);
  conv(s2w[2], ws2[2],  256,  128,  256,  128);
  conv(s3w[0], ws3[0],  256,  259,  256,  288);
  conv(s3w[1], ws3[1],  512,  256,  512,  256);
  conv(s3w[2], ws3[2], 1024,  512, 1024,  512);
  conv(f1w,    wf1,     512, 1024,  512, 1024);
  conv(f2w,    wf2,     256,  512,  256,  512);
  conv(f3w,    wf3,       6,  256,   16,  256);

  transpose_kernel<<<(B * N + 255) / 256, 256, 0, stream>>>(xyz, pts);

  // Stage 1
  fps_kernel<<<B, 256, (size_t)N * 4 + 256 * 8, stream>>>(pts, N, 512, fidx1, nx1);
  ballq_kernel<<<(B * 512 + 127) / 128, 128, 0, stream>>>(
      pts, nx1, N, 512, 32, 0.04f, B * 512, gidx1);
  sa1_kernel<<<B * 512, 128, 0, stream>>>(
      pts, nx1, gidx1, ws1[0], s1b[0], ws1[1], s1b[1], ws1[2], s1b[2], l1p);

  // Stage 2
  fps_kernel<<<B, 256, (size_t)512 * 4 + 256 * 8, stream>>>(nx1, 512, 128, fidx2, nx2);
  ballq_kernel<<<(B * 128 + 127) / 128, 128, 0, stream>>>(
      nx1, nx2, 512, 128, 64, 0.16f, B * 128, gidx2);
  sa2_kernel<<<B * 128, 128, 0, stream>>>(
      nx1, l1p, nx2, gidx2, ws2[0], s2b[0], ws2[1], s2b[1], ws2[2], s2b[2], l2p);

  // Stage 3 (group all): 259(pad288)->256->512->1024, max over 128 points
  fill3_kernel<<<(B * 128 * 288 + 255) / 256, 256, 0, stream>>>(nx2, l2p, X3);
  mlp_global_kernel<<<B, 256, 0, stream>>>(X3, 288, 128, 288, ws3[0], s3b[0], 256, a31, 256);
  mlp_global_kernel<<<B, 256, 0, stream>>>(a31, 256, 128, 256, ws3[1], s3b[1], 512, a32, 512);
  maxout_global_kernel<<<B, 256, 0, stream>>>(a32, 512, 128, 512, ws3[2], s3b[2], 1024, feat);

  // FC head: feat(32,1024) -> 512 -> 256 -> 6
  convw_kernel<<<(B * 1024 + 255) / 256, 256, 0, stream>>>(feat, featH, 32, 1024, 32, 1024);
  mlp_global_kernel<<<1, 256, 0, stream>>>(featH, 1024, 32, 1024, wf1, f1b, 512, fc1, 512);
  mlp_global_kernel<<<1, 256, 0, stream>>>(fc1, 512, 32, 512, wf2, f2b, 256, fc2, 256);
  fc_head_kernel<<<1, 64, 0, stream>>>(fc2, wf3, f3b, out);
  (void)fidx1; (void)fidx2;
}
